// SelfAttention_86260123173417
// MI455X (gfx1250) — matrix operand order repaired and checked by reading, not compiled, not run on hardware
//
#include <hip/hip_runtime.h>
#include <cstdint>

// ---------------------------------------------------------------------------
// MI455X (gfx1250) fused multi-head self-attention forward.
// wave32, WMMA 16x16x32 f16 -> f32 accumulate, async-LDS double buffering.
// ---------------------------------------------------------------------------

typedef __attribute__((ext_vector_type(16))) _Float16 v16h;
typedef __attribute__((ext_vector_type(8)))  float    v8f;
typedef __attribute__((ext_vector_type(4)))  int      v4i;

#define EMBED 1024
#define HEADS 16
#define HD    64
#define NB    4
#define SEQ   2048   // SQ == SK

#if defined(__gfx1250__) && __has_builtin(__builtin_amdgcn_global_load_async_to_lds_b128)
#define HAS_ASYNC_LDS 1
#else
#define HAS_ASYNC_LDS 0
#endif

// 16-byte global -> LDS copy.  Async (ASYNCcnt-tracked, no VGPR round trip)
// when the toolchain exposes the gfx1250 builtin; plain copy otherwise.
__device__ inline void copy16_to_lds(const void* g, void* l) {
#if HAS_ASYNC_LDS
  __builtin_amdgcn_global_load_async_to_lds_b128(
      (__attribute__((address_space(1))) v4i*)(uintptr_t)g,
      (__attribute__((address_space(3))) v4i*)(unsigned int)(uintptr_t)l,
      0, 0);
#else
  *(uint4*)l = *(const uint4*)g;
#endif
}

template <int N>
__device__ inline void async_wait() {
#if HAS_ASYNC_LDS
#if __has_builtin(__builtin_amdgcn_s_wait_asynccnt)
  __builtin_amdgcn_s_wait_asynccnt(N);
#else
  asm volatile("s_wait_asynccnt %0" ::"i"(N) : "memory");
#endif
#endif
}

// Stage a 64x64 f16 tile (row stride `rstride` halves) into LDS (stride 72).
// 4 x b128 per thread with a 128-thread block.
__device__ inline void stage_tile(const _Float16* __restrict__ g, int rstride,
                                  _Float16* __restrict__ buf, int tid) {
#pragma unroll
  for (int i = tid * 8; i < 64 * 64; i += 1024) {
    const int r = i >> 6, c = i & 63;
    copy16_to_lds(g + (size_t)r * rstride + c, buf + r * 72 + c);
  }
}

// ---- WMMA fragment builders (layouts per CDNA5 ISA 7.12.2) ----------------
// A (16x32 f16): lane m = lane&15, hi = lane>>4.
//   halves 0..7  hold K =  8*hi + 0..7 ; halves 8..15 hold K = 16+8*hi + 0..7
__device__ inline v16h frag_a(const _Float16* __restrict__ row, int hi) {
  v16h a;
#pragma unroll
  for (int i = 0; i < 8; ++i) a[i] = row[8 * hi + i];
#pragma unroll
  for (int i = 0; i < 8; ++i) a[8 + i] = row[16 + 8 * hi + i];
  return a;
}

// B (32x16 f16): lane n = lane&15, hi = lane>>4.
//   halves 0..7  hold K =  8*hi + 0..7 ; halves 8..15 hold K = 16+8*hi + 0..7
__device__ inline v16h frag_b(const _Float16* __restrict__ row, int hi) {
  v16h b;
#pragma unroll
  for (int i = 0; i < 8; ++i) b[i] = row[8 * hi + i];
#pragma unroll
  for (int i = 0; i < 8; ++i) b[8 + i] = row[16 + 8 * hi + i];
  return b;
}

// ---------------------------------------------------------------------------
// GEMM: Y[M,N] = X[M,1024] * W[N,1024]^T + bias[N]
// OUT_MODE 0: Y -> f16 head-split [B, H, S, 64]     (Q, K projections)
// OUT_MODE 1: Y -> f32 row-major  [M, 1024]         (output projection)
// OUT_MODE 2: Y -> f16 transposed [B, H, 64, SEQ]   (V projection)
// Block = 128 threads (4 waves); tile 64x64, K-step 32.
// ---------------------------------------------------------------------------
template <typename TIN, int OUT_MODE>
__global__ __launch_bounds__(128) void gemm_wmma(
    const TIN* __restrict__ X, const float* __restrict__ W,
    const float* __restrict__ bias, void* __restrict__ outp) {
  __shared__ __align__(16) _Float16 As[64][40];  // [m][k]
  __shared__ __align__(16) _Float16 Bs[64][40];  // [n][k] == W^T staged

  const int tid = threadIdx.x;
  const int wave = tid >> 5;
  const int lane = tid & 31;
  const int li = lane & 15, hi = lane >> 4;
  const int mbase = blockIdx.y * 64;
  const int nbase = blockIdx.x * 64;

  v8f acc[4] = {};

  for (int kt = 0; kt < EMBED / 32; ++kt) {
    const int k0 = kt * 32;
    if (kt + 1 < EMBED / 32) {  // prefetch next K-step tiles into caches
      const int pr = tid >> 1, pc = (tid & 1) * 16;
      __builtin_prefetch(X + (size_t)(mbase + pr) * EMBED + k0 + 32 + pc, 0, 0);
      __builtin_prefetch(W + (size_t)(nbase + pr) * EMBED + k0 + 32 + pc, 0, 0);
    }
#pragma unroll
    for (int p = 0; p < 4; ++p) {
      int linear = (p * 128 + tid) * 4;
      int r = linear >> 5, c = linear & 31;
      {
        const TIN* g = X + (size_t)(mbase + r) * EMBED + k0 + c;
        _Float16* s = &As[r][c];
        s[0] = (_Float16)g[0]; s[1] = (_Float16)g[1];
        s[2] = (_Float16)g[2]; s[3] = (_Float16)g[3];
      }
      {
        const float* g = W + (size_t)(nbase + r) * EMBED + k0 + c;
        _Float16* s = &Bs[r][c];
        s[0] = (_Float16)g[0]; s[1] = (_Float16)g[1];
        s[2] = (_Float16)g[2]; s[3] = (_Float16)g[3];
      }
    }
    __syncthreads();

    v16h a = frag_a(&As[wave * 16 + li][0], hi);
#pragma unroll
    for (int ns = 0; ns < 4; ++ns) {
      v16h b = frag_b(&Bs[ns * 16 + li][0], hi);
      acc[ns] = __builtin_amdgcn_wmma_f32_16x16x32_f16(
          false, a, false, b, (short)0, acc[ns], false, false);
    }
    __syncthreads();
  }

  // Epilogue. C layout: vgpr r -> row r + 8*hi, col = lane&15.
  if (OUT_MODE == 2) {
    // V transposed: lane's 8 accumulator rows = 8 consecutive s -> one b128.
    const int b = mbase >> 11;
    const int sbase = (mbase & (SEQ - 1)) + wave * 16 + 8 * hi;
#pragma unroll
    for (int ns = 0; ns < 4; ++ns) {
      const int n = nbase + ns * 16 + li;
      const float bv = bias[n];
      const int h = n >> 6, d = n & 63;
      union { uint4 u; _Float16 hh[8]; } t;
#pragma unroll
      for (int r = 0; r < 8; ++r) t.hh[r] = (_Float16)(acc[ns][r] + bv);
      *(uint4*)((_Float16*)outp +
                (((size_t)b * HEADS + h) * HD + d) * SEQ + sbase) = t.u;
    }
  } else {
#pragma unroll
    for (int ns = 0; ns < 4; ++ns) {
      const int n = nbase + ns * 16 + li;
      const float bv = bias[n];
#pragma unroll
      for (int r = 0; r < 8; ++r) {
        const int m = mbase + wave * 16 + r + 8 * hi;
        const float v = acc[ns][r] + bv;
        if (OUT_MODE == 0) {
          const int b = m >> 11, s = m & (SEQ - 1);
          const int h = n >> 6, d = n & 63;
          ((_Float16*)outp)[(((size_t)b * HEADS + h) * SEQ + s) * HD + d] =
              (_Float16)v;
        } else {
          ((float*)outp)[(size_t)m * EMBED + n] = v;
        }
      }
    }
  }
}

// ---------------------------------------------------------------------------
// Attention: per (b, h, 64-row q tile).  Two-pass online softmax with
// async-LDS double-buffered K (pass 1) and K+V (pass 2) tiles.
//   pass 1: S = Q K^T * scale, track row max m / row sum l (never stored).
//   pass 2: recompute S, P = exp(S-m)/l -> NT-store attn_weights (1 GB,
//           streamed past L2), stage P in LDS, accumulate O += P V.
// Block = 128 threads (4 waves); each wave owns 16 q rows.
// ---------------------------------------------------------------------------
__global__ __launch_bounds__(128) void attention_wmma(
    const _Float16* __restrict__ Qh, const _Float16* __restrict__ Kh,
    const _Float16* __restrict__ Vh, float* __restrict__ attnw,
    _Float16* __restrict__ ctx) {
  __shared__ __align__(16) _Float16 Qs[64][72];     // [q][d]
  __shared__ __align__(16) _Float16 Ks[2][64][72];  // [kcol][d] double-buffer
  __shared__ __align__(16) _Float16 Vs[2][64][72];  // [d][kcol] double-buffer
  __shared__ __align__(16) _Float16 Ps[64][72];     // [q][kcol]

  const int tid = threadIdx.x;
  const int wave = tid >> 5;
  const int lane = tid & 31;
  const int li = lane & 15, hi = lane >> 4;
  const int qt = blockIdx.x, h = blockIdx.y, b = blockIdx.z;
  const size_t headbase = ((size_t)b * HEADS + h) * (size_t)SEQ * HD;
  const _Float16* Kb = Kh + headbase;        // [SEQ][HD]
  const _Float16* Vb = Vh + headbase;        // [HD][SEQ] (pre-transposed)
  const int qbase = qt * 64;
  const float scale = 0.125f;  // 1/sqrt(64)

  // stage Q tile (async), wait, barrier
  stage_tile(Qh + headbase + (size_t)qbase * HD, HD, &Qs[0][0], tid);
  async_wait<0>();
  __syncthreads();

  v16h qa[2];
#pragma unroll
  for (int c = 0; c < 2; ++c) qa[c] = frag_a(&Qs[wave * 16 + li][c * 32], hi);

  float mrow[8], lrow[8];
#pragma unroll
  for (int r = 0; r < 8; ++r) { mrow[r] = -1e30f; lrow[r] = 0.f; }

  // ---------------- pass 1: row statistics ----------------
  stage_tile(Kb, HD, &Ks[0][0][0], tid);  // preload tile 0 (4 async/thread)
  for (int kt = 0; kt < SEQ / 64; ++kt) {
    const int cur = kt & 1;
    if (kt + 1 < SEQ / 64) {
      stage_tile(Kb + (size_t)(kt + 1) * 64 * HD, HD, &Ks[cur ^ 1][0][0], tid);
      async_wait<4>();   // only the 4 just-issued may remain outstanding
    } else {
      async_wait<0>();
    }
    __syncthreads();  // tile kt visible to whole workgroup

    v8f s[4];
#pragma unroll
    for (int ns = 0; ns < 4; ++ns) {
      v8f acc = {};
#pragma unroll
      for (int c = 0; c < 2; ++c) {
        v16h bf = frag_b(&Ks[cur][ns * 16 + li][c * 32], hi);
        acc = __builtin_amdgcn_wmma_f32_16x16x32_f16(
            false, qa[c], false, bf, (short)0, acc, false, false);
      }
      s[ns] = acc;
    }

    float mloc[8];
#pragma unroll
    for (int r = 0; r < 8; ++r) {
      float v = fmaxf(fmaxf(s[0][r], s[1][r]), fmaxf(s[2][r], s[3][r]));
      mloc[r] = v * scale;
    }
#pragma unroll
    for (int off = 1; off < 16; off <<= 1)
#pragma unroll
      for (int r = 0; r < 8; ++r)
        mloc[r] = fmaxf(mloc[r], __shfl_xor(mloc[r], off, 32));

    float mnew[8], ps[8];
#pragma unroll
    for (int r = 0; r < 8; ++r) {
      mnew[r] = fmaxf(mrow[r], mloc[r]);
      float p = 0.f;
#pragma unroll
      for (int ns = 0; ns < 4; ++ns) p += __expf(s[ns][r] * scale - mnew[r]);
      ps[r] = p;
    }
#pragma unroll
    for (int off = 1; off < 16; off <<= 1)
#pragma unroll
      for (int r = 0; r < 8; ++r) ps[r] += __shfl_xor(ps[r], off, 32);
#pragma unroll
    for (int r = 0; r < 8; ++r) {
      lrow[r] = lrow[r] * __expf(mrow[r] - mnew[r]) + ps[r];
      mrow[r] = mnew[r];
    }
    __syncthreads();  // everyone done reading Ks[cur] before it is re-filled
  }

  float rinv[8];
#pragma unroll
  for (int r = 0; r < 8; ++r) rinv[r] = 1.f / lrow[r];

  // ---------------- pass 2: P write + P@V ----------------
  v8f o[4] = {};
  const size_t attnbase = ((size_t)b * HEADS + h) * (size_t)SEQ * SEQ;
  stage_tile(Kb, HD, &Ks[0][0][0], tid);   // preload tile-pair 0
  stage_tile(Vb, SEQ, &Vs[0][0][0], tid);  // (8 async/thread)
  for (int kt = 0; kt < SEQ / 64; ++kt) {
    const int cur = kt & 1;
    if (kt + 1 < SEQ / 64) {
      stage_tile(Kb + (size_t)(kt + 1) * 64 * HD, HD, &Ks[cur ^ 1][0][0], tid);
      stage_tile(Vb + (size_t)(kt + 1) * 64, SEQ, &Vs[cur ^ 1][0][0], tid);
      async_wait<8>();
    } else {
      async_wait<0>();
    }
    __syncthreads();

#pragma unroll
    for (int ns = 0; ns < 4; ++ns) {
      v8f acc = {};
#pragma unroll
      for (int c = 0; c < 2; ++c) {
        v16h bf = frag_b(&Ks[cur][ns * 16 + li][c * 32], hi);
        acc = __builtin_amdgcn_wmma_f32_16x16x32_f16(
            false, qa[c], false, bf, (short)0, acc, false, false);
      }
      const int n = kt * 64 + ns * 16 + li;
#pragma unroll
      for (int r = 0; r < 8; ++r) {
        const int qrow = wave * 16 + r + 8 * hi;
        const float p = __expf(acc[r] * scale - mrow[r]) * rinv[r];
        // 1 GB streaming output: non-temporal, keep L2 for K/V tiles
        __builtin_nontemporal_store(
            p, &attnw[attnbase + (size_t)(qbase + qrow) * SEQ + n]);
        Ps[qrow][ns * 16 + li] = (_Float16)p;
      }
    }
    __syncthreads();  // Ps visible

    v16h pa[2];
#pragma unroll
    for (int c = 0; c < 2; ++c)
      pa[c] = frag_a(&Ps[wave * 16 + li][c * 32], hi);
#pragma unroll
    for (int ds = 0; ds < 4; ++ds) {
#pragma unroll
      for (int c = 0; c < 2; ++c) {
        v16h bf = frag_b(&Vs[cur][ds * 16 + li][c * 32], hi);
        o[ds] = __builtin_amdgcn_wmma_f32_16x16x32_f16(
            false, pa[c], false, bf, (short)0, o[ds], false, false);
      }
    }
    __syncthreads();  // done with Ks/Vs[cur] before re-fill
  }

  // context out: [B, SQ, E] f16, interleaved heads
#pragma unroll
  for (int ds = 0; ds < 4; ++ds) {
    const int d = ds * 16 + li;
#pragma unroll
    for (int r = 0; r < 8; ++r) {
      const int q = qbase + wave * 16 + r + 8 * hi;
      ctx[((size_t)b * SEQ + q) * EMBED + h * HD + d] = (_Float16)o[ds][r];
    }
  }
}

// ---------------------------------------------------------------------------
extern "C" void kernel_launch(void* const* d_in, const int* in_sizes, int n_in,
                              void* d_out, int out_size, void* d_ws,
                              size_t ws_size, hipStream_t stream) {
  (void)in_sizes; (void)n_in; (void)out_size; (void)ws_size;
  const float* query = (const float*)d_in[0];
  const float* key   = (const float*)d_in[1];
  const float* value = (const float*)d_in[2];
  const float* Wq = (const float*)d_in[3];
  const float* bq = (const float*)d_in[4];
  const float* Wk = (const float*)d_in[5];
  const float* bk = (const float*)d_in[6];
  const float* Wv = (const float*)d_in[7];
  const float* bv = (const float*)d_in[8];
  const float* Wo = (const float*)d_in[9];
  const float* bo = (const float*)d_in[10];

  // workspace: f16 Q, K (head-split), V (transposed), context.  64 MiB total.
  const size_t perT = (size_t)NB * HEADS * SEQ * HD;  // 8,388,608 halves
  _Float16* Qh  = (_Float16*)d_ws;
  _Float16* Kh  = Qh + perT;
  _Float16* Vh  = Kh + perT;   // [B, H, 64, SEQ]
  _Float16* Ctx = Vh + perT;

  float* out   = (float*)d_out;                   // [B, SQ, E]
  float* attnw = out + (size_t)NB * SEQ * EMBED;  // [B, H, SQ, SK]

  const dim3 gGemm(EMBED / 64, (NB * SEQ) / 64);  // (16, 128)
  gemm_wmma<float, 0><<<gGemm, 128, 0, stream>>>(query, Wq, bq, (void*)Qh);
  gemm_wmma<float, 0><<<gGemm, 128, 0, stream>>>(key,   Wk, bk, (void*)Kh);
  gemm_wmma<float, 2><<<gGemm, 128, 0, stream>>>(value, Wv, bv, (void*)Vh);

  const dim3 gAttn(SEQ / 64, HEADS, NB);          // (32, 16, 4)
  attention_wmma<<<gAttn, 128, 0, stream>>>(Qh, Kh, Vh, attnw, Ctx);

  gemm_wmma<_Float16, 1><<<gGemm, 128, 0, stream>>>(Ctx, Wo, bo, (void*)out);
}
